// LSTMCell_5446018532085
// MI455X (gfx1250) — matrix operand order repaired and checked by reading, not compiled, not run on hardware
//
#include <hip/hip_runtime.h>

typedef __attribute__((ext_vector_type(16))) __bf16 v16bf;
typedef __attribute__((ext_vector_type(8)))  __bf16 v8bf;
typedef __attribute__((ext_vector_type(8)))  float  v8f;
typedef int v4i __attribute__((vector_size(16)));

#define B_DIM 4096
#define K_DIM 2048
#define H_DIM 2048
#define BM 64
#define BN 64
#define BK 32
#define NKT 64          // K_DIM / BK
#define TILE_B 1024     // one 16x32 bf16 fragment tile, 64 chunks of 16B

// workspace byte layout
#define GATES_OFF  ((size_t)0)                    // 4 * B*H fp32 activated gates = 128MB
#define APK_OFF    ((size_t)128 << 20)            // A packed: 4 planes * 16MB = 64MB
#define A_PLANE    ((size_t)16 << 20)             // 256 mt * 64 kt * 1KB
#define BPK_OFF    ((size_t)192 << 20)            // B packed: 16 planes * 8MB = 128MB
#define B_PLANE    ((size_t)8 << 20)              // 128 nt * 64 kt * 1KB

struct GateParams {
  const float* wx[4];
  const float* wh[4];
  const float* bias[4];
};

// ---------------- CDNA5 async copy: global -> LDS (16B per lane) ----------------
__device__ __forceinline__ void async_ld16(void* lds, const void* g) {
#if __has_builtin(__builtin_amdgcn_global_load_async_to_lds_b128)
  __builtin_amdgcn_global_load_async_to_lds_b128(
      (__attribute__((address_space(1))) v4i*)(unsigned long long)(size_t)g,
      (__attribute__((address_space(3))) v4i*)(unsigned)(size_t)lds,
      0, 0);
#else
  *(float4*)lds = *(const float4*)g;   // synchronous fallback
#endif
}

template <int N>
__device__ __forceinline__ void wait_async_le() {
#if __has_builtin(__builtin_amdgcn_s_wait_asynccnt)
  __builtin_amdgcn_s_wait_asynccnt(N);
#else
  asm volatile("s_wait_asynccnt %0" :: "i"(N) : "memory");
#endif
}

// Build a v16bf fragment from two 16B LDS chunks (2x ds_load_b128).
__device__ __forceinline__ v16bf load_frag2(const __bf16* p, int o0, int o1) {
  union { v16bf v; v8bf h[2]; } u;
  u.h[0] = *(const v8bf*)(p + o0);
  u.h[1] = *(const v8bf*)(p + o1);
  return u.v;
}

__device__ __forceinline__ v8f wmma_bf16(v16bf a, v16bf b, v8f c) {
  return __builtin_amdgcn_wmma_f32_16x16x32_bf16(
      false, a, false, b, (short)0, c, false, false);
}

// ---------------- pack kernels: fp32 -> fragment-ordered bf16 hi/lo tiles ----------------
// A-fragment tile (16m x 32k): chunk c: half=c>>5, lane=c&31, lm=lane&15, lh=lane>>4
//   row m = lm, K base = half*16 + lh*8  (8 contiguous K values)
__global__ __launch_bounds__(256) void pack_A(const float* __restrict__ x,
                                              const float* __restrict__ h0,
                                              unsigned char* __restrict__ ws) {
  const unsigned id = blockIdx.x * 256 + threadIdx.x;   // 2^21 threads
  const int c  = id & 63;
  const int kt = (id >> 6) & 63;
  const int mt = (id >> 12) & 255;
  const int s  = (id >> 20) & 1;
  const float* __restrict__ src = s ? h0 : x;
  const int half = c >> 5, lane = c & 31, lm = lane & 15, lh = lane >> 4;
  const int m  = mt * 16 + lm;
  const int kb = kt * BK + half * 16 + lh * 8;
  const float* p = src + (size_t)m * K_DIM + kb;
  const float4 v0 = *(const float4*)p;
  const float4 v1 = *(const float4*)(p + 4);
  const float vals[8] = {v0.x, v0.y, v0.z, v0.w, v1.x, v1.y, v1.z, v1.w};
  v8bf hi, lo;
#pragma unroll
  for (int e = 0; e < 8; ++e) {
    __bf16 h = (__bf16)vals[e];
    hi[e] = h;
    lo[e] = (__bf16)(vals[e] - (float)h);
  }
  const size_t tile = ((size_t)mt * NKT + kt) * TILE_B + (size_t)c * 16;
  *(v8bf*)(ws + APK_OFF + (size_t)(s * 2 + 0) * A_PLANE + tile) = hi;
  *(v8bf*)(ws + APK_OFF + (size_t)(s * 2 + 1) * A_PLANE + tile) = lo;
}

// B-fragment tile (32k x 16n): chunk c: col n = lm, K base = half*16 + lh*8
__global__ __launch_bounds__(256) void pack_B(GateParams P,
                                              unsigned char* __restrict__ ws) {
  const unsigned id = blockIdx.x * 256 + threadIdx.x;   // 2^22 threads
  const int c  = id & 63;
  const int kt = (id >> 6) & 63;
  const int nt = (id >> 12) & 127;
  const int g  = (id >> 19) & 3;
  const int t  = (id >> 21) & 1;
  const float* __restrict__ src = t ? P.wh[g] : P.wx[g];
  const int half = c >> 5, lane = c & 31, lm = lane & 15, lh = lane >> 4;
  const int n  = nt * 16 + lm;
  const int kb = kt * BK + half * 16 + lh * 8;
  float vals[8];
  if (t == 0) {
    // x-weights used as w_x^T: column n of B == row n of w_x, K-contiguous
    const float* p = src + (size_t)n * K_DIM + kb;
    const float4 v0 = *(const float4*)p;
    const float4 v1 = *(const float4*)(p + 4);
    vals[0]=v0.x; vals[1]=v0.y; vals[2]=v0.z; vals[3]=v0.w;
    vals[4]=v1.x; vals[5]=v1.y; vals[6]=v1.z; vals[7]=v1.w;
  } else {
    // h-weights used directly: B[k][n] = w_h[k][n], K-strided (one-time transpose cost)
#pragma unroll
    for (int e = 0; e < 8; ++e) vals[e] = src[(size_t)(kb + e) * H_DIM + n];
  }
  v8bf hi, lo;
#pragma unroll
  for (int e = 0; e < 8; ++e) {
    __bf16 h = (__bf16)vals[e];
    hi[e] = h;
    lo[e] = (__bf16)(vals[e] - (float)h);
  }
  const size_t tile = ((size_t)nt * NKT + kt) * TILE_B + (size_t)c * 16;
  const size_t pl = (size_t)(t * 4 + g) * 2;
  *(v8bf*)(ws + BPK_OFF + (pl + 0) * B_PLANE + tile) = hi;
  *(v8bf*)(ws + BPK_OFF + (pl + 1) * B_PLANE + tile) = lo;
}

// ---------------- main GEMM: 4 gates, split-bf16 fp32 emulation ----------------
__global__ __launch_bounds__(128) void lstm_gates_gemm(
    GateParams P, unsigned char* __restrict__ ws) {

  // double-buffered staging: per buffer 8 planes x 4 subtiles x 1KB = 32KB
  __shared__ __align__(16) unsigned char sbuf[2][32768];

  const int tid  = threadIdx.x;
  const int g    = blockIdx.z;
  const int bm16 = blockIdx.x * 4;   // first m-subtile index
  const int bn16 = blockIdx.y * 4;   // first n-subtile index

  const int lane = tid & 31;
  const int wave = tid >> 5;
  const int wr   = wave >> 1;
  const int wc   = wave & 1;
  const int lm   = lane & 15;
  const int lh   = lane >> 4;

  const unsigned char* apk = ws + APK_OFF;
  const unsigned char* bpk = ws + BPK_OFF;

  v8f acc[2][2];
  acc[0][0] = {}; acc[0][1] = {}; acc[1][0] = {}; acc[1][1] = {};

  // stage one k-step: 16 async b128 per thread (4 A planes + 4 B planes, wave = subtile)
  auto stage = [&](unsigned char* sb, int kt) {
#pragma unroll
    for (int p = 0; p < 4; ++p) {
      const unsigned char* gs =
          apk + (size_t)p * A_PLANE + ((size_t)(bm16 + wave) * NKT + kt) * TILE_B;
      unsigned char* ld = sb + p * 4096 + wave * TILE_B;
      async_ld16(ld + lane * 16,       gs + lane * 16);
      async_ld16(ld + 512 + lane * 16, gs + 512 + lane * 16);
    }
#pragma unroll
    for (int p = 0; p < 4; ++p) {
      const size_t gp = (p < 2) ? (size_t)(g * 2 + p) : (size_t)(8 + g * 2 + (p - 2));
      const unsigned char* gs =
          bpk + gp * B_PLANE + ((size_t)(bn16 + wave) * NKT + kt) * TILE_B;
      unsigned char* ld = sb + 16384 + p * 4096 + wave * TILE_B;
      async_ld16(ld + lane * 16,       gs + lane * 16);
      async_ld16(ld + 512 + lane * 16, gs + 512 + lane * 16);
    }
  };

  stage(sbuf[0], 0);

  for (int kt = 0; kt < NKT; ++kt) {
    const int cur = kt & 1;
    const bool more = (kt + 1) < NKT;
    if (more) {
      stage(sbuf[cur ^ 1], kt + 1);
      wait_async_le<16>();          // in-order returns: current buffer's 16 are done
    } else {
      wait_async_le<0>();
    }
    __syncthreads();

    const __bf16* sb = (const __bf16*)sbuf[cur];
    const int o0 = lane * 8;        // chunk(half=0): bytes lane*16
    const int o1 = lane * 8 + 256;  // chunk(half=1): bytes 512 + lane*16

    v16bf axh[2], axl[2], ahh[2], ahl[2];
    v16bf bxh[2], bxl[2], bhh[2], bhl[2];
#pragma unroll
    for (int mt = 0; mt < 2; ++mt) {
      const int st = (wr * 2 + mt) * 512;         // subtile offset in bf16 elems
      axh[mt] = load_frag2(sb + 0 * 2048 + st, o0, o1);
      axl[mt] = load_frag2(sb + 1 * 2048 + st, o0, o1);
      ahh[mt] = load_frag2(sb + 2 * 2048 + st, o0, o1);
      ahl[mt] = load_frag2(sb + 3 * 2048 + st, o0, o1);
    }
#pragma unroll
    for (int nt = 0; nt < 2; ++nt) {
      const int st = (wc * 2 + nt) * 512;
      bxh[nt] = load_frag2(sb + 8192 + 0 * 2048 + st, o0, o1);
      bxl[nt] = load_frag2(sb + 8192 + 1 * 2048 + st, o0, o1);
      bhh[nt] = load_frag2(sb + 8192 + 2 * 2048 + st, o0, o1);
      bhl[nt] = load_frag2(sb + 8192 + 3 * 2048 + st, o0, o1);
    }

    // hi*hi + hi*lo + lo*hi for both GEMM streams
#pragma unroll
    for (int mt = 0; mt < 2; ++mt)
#pragma unroll
      for (int nt = 0; nt < 2; ++nt) {
        v8f c = acc[mt][nt];
        c = wmma_bf16(axh[mt], bxh[nt], c);
        c = wmma_bf16(axh[mt], bxl[nt], c);
        c = wmma_bf16(axl[mt], bxh[nt], c);
        c = wmma_bf16(ahh[mt], bhh[nt], c);
        c = wmma_bf16(ahh[mt], bhl[nt], c);
        c = wmma_bf16(ahl[mt], bhh[nt], c);
        acc[mt][nt] = c;
      }
    __syncthreads();
  }

  // epilogue: bias + activation -> activated gate plane in workspace
  float* __restrict__ plane = (float*)(ws + GATES_OFF) + (size_t)g * B_DIM * H_DIM;
  const float* __restrict__ bias = P.bias[g];
  const int bm = blockIdx.x * BM;
  const int bn = blockIdx.y * BN;
#pragma unroll
  for (int mt = 0; mt < 2; ++mt)
#pragma unroll
    for (int nt = 0; nt < 2; ++nt) {
      const int col = bn + wc * 32 + nt * 16 + lm;
      const float bb = bias[col];
#pragma unroll
      for (int r = 0; r < 8; ++r) {
        const int row = bm + wr * 32 + mt * 16 + lh * 8 + r;
        float v = acc[mt][nt][r] + bb;
        v = (g == 3) ? tanhf(v) : (1.0f / (1.0f + expf(-v)));
        plane[(size_t)row * H_DIM + col] = v;
      }
    }
}

// ---------------- fuse + LayerNorm ----------------
__global__ __launch_bounds__(256) void lstm_fuse_ln(
    const float* __restrict__ ws, const float* __restrict__ c0,
    float* __restrict__ out) {
  const size_t BH = (size_t)B_DIM * H_DIM;
  const int row = blockIdx.x;
  const int tid = threadIdx.x;
  const float* __restrict__ gi = ws;
  const float* __restrict__ gf = ws + BH;
  const float* __restrict__ go = ws + 2 * BH;
  const float* __restrict__ gc = ws + 3 * BH;

  float tv[8], ov[8];
  float s1 = 0.0f, s2 = 0.0f;
#pragma unroll
  for (int t = 0; t < 8; ++t) {
    const size_t idx = (size_t)row * H_DIM + tid + t * 256;
    const float iv = gi[idx], fv = gf[idx], o = go[idx], cv = gc[idx], cz = c0[idx];
    const float tt = fv * cz + iv * cv;
    tv[t] = tt; ov[t] = o;
    s1 += tt; s2 += tt * tt;
  }

  __shared__ float r1[256], r2[256];
  r1[tid] = s1; r2[tid] = s2;
  __syncthreads();
#pragma unroll
  for (int s = 128; s > 0; s >>= 1) {
    if (tid < s) { r1[tid] += r1[tid + s]; r2[tid] += r2[tid + s]; }
    __syncthreads();
  }
  const float mean = r1[0] * (1.0f / H_DIM);
  const float var  = r2[0] * (1.0f / H_DIM) - mean * mean;
  const float rinv = rsqrtf(var + 1e-5f);

#pragma unroll
  for (int t = 0; t < 8; ++t) {
    const size_t idx = (size_t)row * H_DIM + tid + t * 256;
    const float c = (tv[t] - mean) * rinv;
    out[idx]      = ov[t] * tanhf(c);   // h plane first
    out[BH + idx] = c;                  // then c plane
  }
}

extern "C" void kernel_launch(void* const* d_in, const int* in_sizes, int n_in,
                              void* d_out, int out_size, void* d_ws, size_t ws_size,
                              hipStream_t stream) {
  (void)in_sizes; (void)n_in; (void)out_size; (void)ws_size;
  const float* x  = (const float*)d_in[0];
  const float* h0 = (const float*)d_in[1];
  const float* c0 = (const float*)d_in[2];

  GateParams P;
  P.wx[0] = (const float*)d_in[3];  P.wh[0] = (const float*)d_in[4];  P.bias[0] = (const float*)d_in[5];
  P.wx[1] = (const float*)d_in[6];  P.wh[1] = (const float*)d_in[7];  P.bias[1] = (const float*)d_in[8];
  P.wx[2] = (const float*)d_in[9];  P.wh[2] = (const float*)d_in[10]; P.bias[2] = (const float*)d_in[11];
  P.wx[3] = (const float*)d_in[12]; P.wh[3] = (const float*)d_in[13]; P.bias[3] = (const float*)d_in[14];

  unsigned char* wsb = (unsigned char*)d_ws;
  float* out = (float*)d_out;

  pack_A<<<8192, 256, 0, stream>>>(x, h0, wsb);
  pack_B<<<16384, 256, 0, stream>>>(P, wsb);
  lstm_gates_gemm<<<dim3(B_DIM / BM, H_DIM / BN, 4), 128, 0, stream>>>(P, wsb);
  lstm_fuse_ln<<<B_DIM, 256, 0, stream>>>((const float*)wsb, c0, out);
}
